// QuantLinear_6451040879067
// MI455X (gfx1250) — compile-verified, matrix operand order checked by reading, not run on hardware
//
#include <hip/hip_runtime.h>

// -------- CDNA5 WMMA types --------
typedef __attribute__((ext_vector_type(16))) __bf16 v16bf;
typedef __attribute__((ext_vector_type(8)))  __bf16 v8bf;
typedef __attribute__((ext_vector_type(4)))  __bf16 v4bf;
typedef __attribute__((ext_vector_type(8)))  float  v8f;

// Types for the gfx1250 async-to-LDS builtin:
// param0: 'int __vector(4) __device__ *'  (AS1, non-const)
// param1: LDS destination (AS3)
typedef int vint4_t __attribute__((__vector_size__(16)));
typedef __attribute__((address_space(1))) vint4_t* gvi4p;
typedef __attribute__((address_space(3))) vint4_t* lvi4p;

#define K_DIM 4096
#define N_DIM 11008
#define M_DIM 8192

#define BM 128            // output tile rows per block
#define BN 128            // output tile cols per block
#define KB 64             // K elements (fp32/int32) staged per chunk
#define LDSTRIDE 72       // bf16 elements per LDS row (64 + 8 pad)
#define RAWSTRIDE 68      // int32 elements per raw LDS row (64 + 4 pad -> 272B)
#define NCHUNK (K_DIM / KB)

// y[m,n] = (sum_k x[m,k]*wq[n,k]) * scale[n] + bias[n]
//
//  - memory-bound (~675MB @ 23.3TB/s ~= 29us); compute headroom ~30x.
//  - activations: bf16x2 split (hi + residual lo) -> ~fp32-grade dot products.
//  - weights: streamed via gfx1250 GLOBAL_LOAD_ASYNC_TO_LDS (ASYNCcnt-tracked
//    DMA, no VGPR staging); int8 values convert EXACTLY to bf16 at fragment
//    build. Per-channel scale/bias applied on the f32 accumulator.
__global__ __launch_bounds__(256)
void qlinear_wmma_bf16x2_async(const float* __restrict__ x,
                               const int*   __restrict__ wq,
                               const float* __restrict__ scale,
                               const float* __restrict__ bias,
                               float* __restrict__ out) {
  __shared__ __bf16 Ah[2][BM * LDSTRIDE];    // x tile, hi bf16 plane
  __shared__ __bf16 Al[2][BM * LDSTRIDE];    // x tile, lo (residual) plane
  __shared__ int    Braw[2][BN * RAWSTRIDE]; // weight tile, raw int32 via async DMA

  const int tid  = threadIdx.x;
  const int lane = tid & 31;
  const int wave = tid >> 5;
  const int wm   = wave & 3;     // 4 waves along M: 32 rows each
  const int wn   = wave >> 2;    // 2 waves along N: 64 cols each

  const int m_base = blockIdx.y * BM;
  const int n_base = blockIdx.x * BN;

  // Staging map: 256 threads = 16 rows x 16 (4-elem groups); 8 passes = 128 rows.
  const int lrow = tid >> 4;     // 0..15
  const int lcol = tid & 15;     // 4-element column group

  const float* xg = x  + (size_t)m_base * K_DIM + lcol * 4;
  const int*   wg = wq + (size_t)n_base * K_DIM + lcol * 4;

  float4 areg[8];

  // ---- prologue: async-DMA weight chunk 0 into LDS, load x chunk 0 ----
  #pragma unroll
  for (int p = 0; p < 8; ++p) {
    const int row = lrow + p * 16;
    __builtin_amdgcn_global_load_async_to_lds_b128(
        (gvi4p)(vint4_t*)const_cast<int*>(wg + (size_t)row * K_DIM),
        (lvi4p)(void*)(&Braw[0][row * RAWSTRIDE + lcol * 4]),
        0, 0);
    areg[p] = *reinterpret_cast<const float4*>(xg + (size_t)row * K_DIM);
  }

  v8f acc[2][4];
  #pragma unroll
  for (int mt = 0; mt < 2; ++mt)
    #pragma unroll
    for (int nt = 0; nt < 4; ++nt)
      acc[mt][nt] = (v8f){0.f, 0.f, 0.f, 0.f, 0.f, 0.f, 0.f, 0.f};

  const int half = lane >> 4;    // 0: lanes 0-15, 1: lanes 16-31
  const int lr   = lane & 15;

  int buf = 0;
  for (int c = 0; c < NCHUNK; ++c) {
    // ---- convert staged x regs -> bf16 hi/lo planes in LDS[buf] ----
    #pragma unroll
    for (int p = 0; p < 8; ++p) {
      const int row = lrow + p * 16;
      const int off = row * LDSTRIDE + lcol * 4;
      v4bf hv, lv;
      float f0 = areg[p].x, f1 = areg[p].y, f2 = areg[p].z, f3 = areg[p].w;
      hv[0] = (__bf16)f0; hv[1] = (__bf16)f1; hv[2] = (__bf16)f2; hv[3] = (__bf16)f3;
      lv[0] = (__bf16)(f0 - (float)hv[0]);
      lv[1] = (__bf16)(f1 - (float)hv[1]);
      lv[2] = (__bf16)(f2 - (float)hv[2]);
      lv[3] = (__bf16)(f3 - (float)hv[3]);
      *reinterpret_cast<v4bf*>(&Ah[buf][off]) = hv;
      *reinterpret_cast<v4bf*>(&Al[buf][off]) = lv;
    }

    // own async transfers for chunk c must have landed before the barrier
    asm volatile("s_wait_asynccnt 0x0" ::: "memory");
    __syncthreads();

    // ---- issue chunk c+1: weight DMA -> LDS[buf^1], x loads -> regs ----
    if (c + 1 < NCHUNK) {
      const int k0 = (c + 1) * KB;
      #pragma unroll
      for (int p = 0; p < 8; ++p) {
        const int row = lrow + p * 16;
        __builtin_amdgcn_global_load_async_to_lds_b128(
            (gvi4p)(vint4_t*)const_cast<int*>(wg + (size_t)row * K_DIM + k0),
            (lvi4p)(void*)(&Braw[buf ^ 1][row * RAWSTRIDE + lcol * 4]),
            0, 0);
        areg[p] = *reinterpret_cast<const float4*>(xg + (size_t)row * K_DIM + k0);
      }
    }

    // ---- compute: 2 k-steps of 32, 16 WMMAs each (hi + lo) ----
    #pragma unroll
    for (int ks = 0; ks < 2; ++ks) {
      const int k1 = ks * 32 + half * 8;       // lane's first K octet
      v16bf ah[2], al[2], bfrag[4];

      // A fragments: lane<16 -> K {k..k+7, k+16..k+23}; lane>=16 shifts K by 8.
      #pragma unroll
      for (int mt = 0; mt < 2; ++mt) {
        const int roff = (wm * 32 + mt * 16 + lr) * LDSTRIDE;
        {
          const __bf16* rp = &Ah[buf][roff];
          v8bf lo = *reinterpret_cast<const v8bf*>(rp + k1);
          v8bf hi = *reinterpret_cast<const v8bf*>(rp + k1 + 16);
          ah[mt] = __builtin_shufflevector(lo, hi,
              0,1,2,3,4,5,6,7,8,9,10,11,12,13,14,15);
        }
        {
          const __bf16* rp = &Al[buf][roff];
          v8bf lo = *reinterpret_cast<const v8bf*>(rp + k1);
          v8bf hi = *reinterpret_cast<const v8bf*>(rp + k1 + 16);
          al[mt] = __builtin_shufflevector(lo, hi,
              0,1,2,3,4,5,6,7,8,9,10,11,12,13,14,15);
        }
      }
      // B fragments: raw int32 from async-DMA'd LDS, exact convert to bf16.
      #pragma unroll
      for (int nt = 0; nt < 4; ++nt) {
        const int* rp = &Braw[buf][(wn * 64 + nt * 16 + lr) * RAWSTRIDE];
        int4 p0 = *reinterpret_cast<const int4*>(rp + k1);
        int4 p1 = *reinterpret_cast<const int4*>(rp + k1 + 4);
        int4 p2 = *reinterpret_cast<const int4*>(rp + k1 + 16);
        int4 p3 = *reinterpret_cast<const int4*>(rp + k1 + 20);
        v16bf b;
        b[0]  = (__bf16)(float)p0.x; b[1]  = (__bf16)(float)p0.y;
        b[2]  = (__bf16)(float)p0.z; b[3]  = (__bf16)(float)p0.w;
        b[4]  = (__bf16)(float)p1.x; b[5]  = (__bf16)(float)p1.y;
        b[6]  = (__bf16)(float)p1.z; b[7]  = (__bf16)(float)p1.w;
        b[8]  = (__bf16)(float)p2.x; b[9]  = (__bf16)(float)p2.y;
        b[10] = (__bf16)(float)p2.z; b[11] = (__bf16)(float)p2.w;
        b[12] = (__bf16)(float)p3.x; b[13] = (__bf16)(float)p3.y;
        b[14] = (__bf16)(float)p3.z; b[15] = (__bf16)(float)p3.w;
        bfrag[nt] = b;
      }

      #pragma unroll
      for (int mt = 0; mt < 2; ++mt)
        #pragma unroll
        for (int nt = 0; nt < 4; ++nt) {
          acc[mt][nt] = __builtin_amdgcn_wmma_f32_16x16x32_bf16(
              false, ah[mt], false, bfrag[nt],
              (short)0, acc[mt][nt], false, false);
          acc[mt][nt] = __builtin_amdgcn_wmma_f32_16x16x32_bf16(
              false, al[mt], false, bfrag[nt],
              (short)0, acc[mt][nt], false, false);
        }
    }

    buf ^= 1;
    __syncthreads();
  }

  // ---- epilogue: dequant scale + bias on f32 accumulators ----
  // C/D layout: VGPR r, lanes 0-15 -> (M=r, N=lane); lanes 16-31 -> (M=r+8, N=lane-16)
  #pragma unroll
  for (int nt = 0; nt < 4; ++nt) {
    const int n = n_base + wn * 64 + nt * 16 + lr;
    const float s = scale[n];
    const float b = bias[n];
    #pragma unroll
    for (int mt = 0; mt < 2; ++mt) {
      const int mrow = m_base + wm * 32 + mt * 16 + half * 8;
      #pragma unroll
      for (int r = 0; r < 8; ++r) {
        out[(size_t)(mrow + r) * N_DIM + n] = acc[mt][nt][r] * s + b;
      }
    }
  }
}

extern "C" void kernel_launch(void* const* d_in, const int* in_sizes, int n_in,
                              void* d_out, int out_size, void* d_ws, size_t ws_size,
                              hipStream_t stream) {
  const float* x     = (const float*)d_in[0];
  const int*   wq    = (const int*)  d_in[1];
  const float* scale = (const float*)d_in[2];
  const float* bias  = (const float*)d_in[3];
  float*       out   = (float*)d_out;

  dim3 grid(N_DIM / BN, M_DIM / BM);   // 86 x 64
  dim3 block(256);
  hipLaunchKernelGGL(qlinear_wmma_bf16x2_async, grid, block, 0, stream,
                     x, wq, scale, bias, out);
}
